// Conv2d_2723009266574
// MI455X (gfx1250) — compile-verified, matrix operand order checked by reading, not run on hardware
//
#include <hip/hip_runtime.h>
#include <hip/hip_bf16.h>
#include <stdint.h>

// ---------------- problem constants (fixed by the reference) ----------------
#define CIN   128
#define COUT  256
#define HH    56
#define WW    56
#define HP    58                 // H + 2 halo
#define WP    58                 // W + 2 halo
#define BATCH 32
#define NPIX  (HH * WW)          // 3136
#define NTOT  (BATCH * NPIX)     // 100352
#define KTOT  (CIN * 9)          // 1152, k = (kh*3+kw)*128 + ci
#define XHELEMS ((size_t)BATCH * HP * WP * CIN)  // 13778944 (halo NHWC)
#define WELEMS  ((size_t)COUT * CIN * 9)         // 294912

// ---------------- WMMA vector types ----------------
typedef __attribute__((ext_vector_type(16))) __bf16 v16bf;
typedef __attribute__((ext_vector_type(8)))  float  v8f;

union AFrag { uint4 u[2]; v16bf v; };
union CTile { v8f v; float f[8]; };

// ---------------- async global->LDS support (CDNA5) ----------------
// Signature (from clang diagnostic): (int4vec global*, int4vec shared*, imm int, imm int)
typedef int vi4_t __attribute__((vector_size(16)));

#if __has_builtin(__builtin_amdgcn_global_load_async_to_lds_b128)
#define HAVE_ASYNC 1
#define GAS(p) ((__attribute__((address_space(1))) vi4_t*)(p))
#define LAS(p) ((__attribute__((address_space(3))) vi4_t*)(p))
#else
#define HAVE_ASYNC 0
#endif

__device__ __forceinline__ void async_wait0() {
#if __has_builtin(__builtin_amdgcn_s_wait_asynccnt)
    __builtin_amdgcn_s_wait_asynccnt(0);
#else
    asm volatile("s_wait_asynccnt 0x0" ::: "memory");
#endif
}

// fp32 -> bf16, round to nearest even
__device__ __forceinline__ unsigned short f2bf(float f) {
    unsigned int u = __float_as_uint(f);
    unsigned int r = u + 0x7FFFu + ((u >> 16) & 1u);
    return (unsigned short)(r >> 16);
}

// ---- pack x (NCHW f32) -> xh (NHWC bf16 with 1-pixel zero halo) ----
__global__ __launch_bounds__(256) void pack_x_halo_bf16(const float* __restrict__ x,
                                                        unsigned short* __restrict__ xh) {
    int tid = blockIdx.x * 256 + threadIdx.x;     // [b][hh][ww][ci] linear
    int ci = tid & 127;
    int t  = tid >> 7;
    int ww = t % WP; t /= WP;
    int hh = t % HP; t /= HP;
    int b  = t;
    float v = 0.0f;
    if (hh >= 1 && hh <= HH && ww >= 1 && ww <= WW) {
        v = x[(((size_t)b * CIN + ci) * HH + (hh - 1)) * WW + (ww - 1)];
    }
    xh[tid] = f2bf(v);
}

// ---- pack W (OIHW f32) -> Wp[co][rs][ci] bf16 ----
__global__ __launch_bounds__(256) void pack_w_bf16(const float* __restrict__ W,
                                                   unsigned short* __restrict__ Wp) {
    int tid = blockIdx.x * 256 + threadIdx.x;     // co*1152 + rs*128 + ci
    int ci = tid & 127;
    int t  = tid >> 7;
    int rs = t % 9;
    int co = t / 9;
    float v = W[((size_t)co * CIN + ci) * 9 + rs];
    Wp[tid] = f2bf(v);
}

// ---------------- implicit-GEMM conv: 128x128 block tile, K chunk 32 ----------------
#define BK   32
#define LSTR 40   // LDS row stride in halves (80B = 20 dwords -> conflict-free b128 reads)

__global__ __launch_bounds__(256) void conv_gemm_wmma(const unsigned short* __restrict__ Wp,
                                                      const unsigned short* __restrict__ xh,
                                                      const float* __restrict__ bias,
                                                      float* __restrict__ out) {
    __shared__ __align__(16) unsigned short As[2][128 * LSTR];   // [m][k]
    __shared__ __align__(16) unsigned short Bs[2][128 * LSTR];   // [n][k]

    const int tid  = threadIdx.x;
    const int lane = tid & 31;
    const int wave = tid >> 5;
    const int wm   = wave >> 2;                    // 0..1 : 64 rows of M
    const int wn   = wave & 3;                     // 0..3 : 32 cols of N
    const int m0   = blockIdx.y * 128;
    const int n0   = blockIdx.x * 128;

    // staging indices: 2 threads per tile row, 16 halves (32B) each
    const int grow = tid >> 1;                     // 0..127
    const int gk   = (tid & 1) * 16;               // 0 or 16 halves

    // im2col coordinates for this thread's B row (output pixel), halo space
    const int p   = n0 + grow;
    const int pb  = p / NPIX;
    const int phw = p - pb * NPIX;
    const int ph  = phw / WW;                      // halo row = ph + r
    const int pw  = phw - ph * WW;                 // halo col = pw + s

    v8f acc[4][2] = {};

#if !HAVE_ASYNC
    uint4 ra0, ra1, rb0, rb1;
#endif

    // ---- stage one K chunk: global -> LDS ----
    auto issue_stage = [&](int kk, int buf) {
        const unsigned short* ga = Wp + (size_t)(m0 + grow) * KTOT + kk + gk;
        int rs  = kk >> 7;
        int r   = rs / 3, s = rs - r * 3;
        int ci0 = (kk & 127) + gk;
        const unsigned short* gb =
            xh + (((size_t)(pb * HP + ph + r)) * WP + (pw + s)) * CIN + ci0;
#if HAVE_ASYNC
        unsigned short* la = &As[buf][grow * LSTR + gk];
        unsigned short* lb = &Bs[buf][grow * LSTR + gk];
        __builtin_amdgcn_global_load_async_to_lds_b128(GAS(ga),     LAS(la),     0, 0);
        __builtin_amdgcn_global_load_async_to_lds_b128(GAS(ga + 8), LAS(la + 8), 0, 0);
        __builtin_amdgcn_global_load_async_to_lds_b128(GAS(gb),     LAS(lb),     0, 0);
        __builtin_amdgcn_global_load_async_to_lds_b128(GAS(gb + 8), LAS(lb + 8), 0, 0);
        (void)buf;
#else
        ra0 = ((const uint4*)ga)[0];
        ra1 = ((const uint4*)ga)[1];
        rb0 = ((const uint4*)gb)[0];
        rb1 = ((const uint4*)gb)[1];
#endif
    };

#if !HAVE_ASYNC
    auto commit_stage = [&](int buf) {
        uint4* pa = (uint4*)&As[buf][grow * LSTR + gk];
        pa[0] = ra0; pa[1] = ra1;
        uint4* pbp = (uint4*)&Bs[buf][grow * LSTR + gk];
        pbp[0] = rb0; pbp[1] = rb1;
    };
#endif

    const int lrow = lane & 15;
    const int kb   = (lane >> 4) * 8;              // halves: lanes 16-31 take K 8-15 / 24-31

    auto compute = [&](int buf) {
        AFrag a[4], b[2];
        #pragma unroll
        for (int mt = 0; mt < 4; ++mt) {
            const unsigned short* pa = &As[buf][(wm * 64 + mt * 16 + lrow) * LSTR + kb];
            a[mt].u[0] = *((const uint4*)pa);
            a[mt].u[1] = *((const uint4*)(pa + 16));
        }
        #pragma unroll
        for (int nt = 0; nt < 2; ++nt) {
            const unsigned short* pbp = &Bs[buf][(wn * 32 + nt * 16 + lrow) * LSTR + kb];
            b[nt].u[0] = *((const uint4*)pbp);
            b[nt].u[1] = *((const uint4*)(pbp + 16));
        }
        #pragma unroll
        for (int mt = 0; mt < 4; ++mt)
            #pragma unroll
            for (int nt = 0; nt < 2; ++nt)
                acc[mt][nt] = __builtin_amdgcn_wmma_f32_16x16x32_bf16(
                    false, a[mt].v, false, b[nt].v, (short)0, acc[mt][nt], false, false);
    };

    // -------- software-pipelined K loop: 36 chunks of 32 --------
    const int NCHUNK = KTOT / BK;                  // 36

#if HAVE_ASYNC
    issue_stage(0, 0);
    async_wait0();
    __syncthreads();
    for (int t = 0; t < NCHUNK; ++t) {
        int buf = t & 1;
        if (t + 1 < NCHUNK) issue_stage((t + 1) * BK, buf ^ 1);   // async copy next chunk
        compute(buf);
        async_wait0();                                            // our async writes landed
        __syncthreads();                                          // everyone's landed
    }
#else
    issue_stage(0, 0);
    commit_stage(0);
    __syncthreads();
    for (int t = 0; t < NCHUNK; ++t) {
        int buf = t & 1;
        if (t + 1 < NCHUNK) issue_stage((t + 1) * BK, buf ^ 1);
        compute(buf);
        if (t + 1 < NCHUNK) commit_stage(buf ^ 1);
        __syncthreads();
    }
#endif

    // -------- epilogue: fp32 accumulators + bias -> NCHW output --------
    const int mhalf = (lane >> 4) * 8;
    #pragma unroll
    for (int mt = 0; mt < 4; ++mt) {
        #pragma unroll
        for (int nt = 0; nt < 2; ++nt) {
            int pcol = n0 + wn * 32 + nt * 16 + lrow;
            int ob   = pcol / NPIX;
            int ohw  = pcol - ob * NPIX;
            CTile c; c.v = acc[mt][nt];
            #pragma unroll
            for (int r = 0; r < 8; ++r) {
                int co = m0 + wm * 64 + mt * 16 + mhalf + r;
                out[((size_t)ob * COUT + co) * NPIX + ohw] = c.f[r] + bias[co];
            }
        }
    }
}

// ---------------- host entry ----------------
extern "C" void kernel_launch(void* const* d_in, const int* in_sizes, int n_in,
                              void* d_out, int out_size, void* d_ws, size_t ws_size,
                              hipStream_t stream) {
    const float* x  = (const float*)d_in[0];
    const float* W  = (const float*)d_in[1];
    const float* b  = (const float*)d_in[2];
    float* out      = (float*)d_out;

    unsigned short* xh = (unsigned short*)d_ws;           // halo NHWC bf16: 27.5 MB
    unsigned short* Wp = xh + XHELEMS;                    // [co][rs][ci] bf16: 0.59 MB

    pack_x_halo_bf16<<<(int)(XHELEMS / 256), 256, 0, stream>>>(x, xh);
    pack_w_bf16<<<(int)(WELEMS / 256), 256, 0, stream>>>(W, Wp);

    dim3 grid(NTOT / 128, COUT / 128);                    // 784 x 2
    conv_gemm_wmma<<<grid, 256, 0, stream>>>(Wp, xh, b, out);
}
